// GPT2Attention_55783035240961
// MI455X (gfx1250) — matrix operand order repaired and checked by reading, not compiled, not run on hardware
//
#include <hip/hip_runtime.h>
#include <hip/hip_bf16.h>

// ---------------------------------------------------------------------------
// Problem constants (from reference)
// ---------------------------------------------------------------------------
#define BATCH       32
#define HID         1024
#define NHEADS      16
#define HEADDIM     64
#define BLKSZ       16
#define MAX_KV      2048
#define BLK_PER_SEQ 128
#define NUM_BLOCKS  4096
#define ATTN_SCALE  0.125f   // 64^-0.5

#define CHUNK_KEYS  64                       // keys per async K chunk (16 KB)
#define CHUNK_FLTS  (CHUNK_KEYS * HEADDIM)   // 4096 floats
#define OPS_PER_CHUNK 4                      // b128 ops per thread per chunk

typedef __attribute__((ext_vector_type(16))) _Float16 v16h;
typedef __attribute__((ext_vector_type(8)))  float    v8f;

// 16-byte vector type matching the async-LDS builtin's pointee type
typedef int v4i_t __attribute__((vector_size(16)));
typedef __attribute__((address_space(1))) v4i_t* gas_v4i_ptr;   // global
typedef __attribute__((address_space(3))) v4i_t* las_v4i_ptr;   // LDS

// ---- gfx1250 async global->LDS path (guarded; fallback = sync copy) -------
#if defined(__has_builtin)
#  if __has_builtin(__builtin_amdgcn_global_load_async_to_lds_b128)
#    define HAVE_ASYNC_LDS 1
#  endif
#endif
#ifndef HAVE_ASYNC_LDS
#  define HAVE_ASYNC_LDS 0
#endif

#if defined(__has_builtin)
#  if __has_builtin(__builtin_amdgcn_s_wait_asynccnt)
#    define ASYNC_WAIT(n) __builtin_amdgcn_s_wait_asynccnt(n)
#  endif
#endif
#ifndef ASYNC_WAIT
#  define ASYNC_WAIT(n) asm volatile("s_wait_asynccnt " #n ::: "memory")
#endif

// ---------------------------------------------------------------------------
// GEMM: Y[M=32, N] = X[32, K] @ W[K, N-slice] + bias  (f16 WMMA, f32 accum)
// One wave computes one 16x16 tile; K unrolled x2 -> 2 v_wmma in flight.
// ---------------------------------------------------------------------------
__global__ __launch_bounds__(32)
void gemm_wmma_f16(const float* __restrict__ X, int ldx,
                   const float* __restrict__ W, int ldw,
                   const float* __restrict__ bias,
                   float* __restrict__ Y, int ldy,
                   int K)
{
    const int lane = threadIdx.x;            // 0..31
    const int n0   = blockIdx.x * 16;
    const int m0   = blockIdx.y * 16;

    const int rowA   = m0 + (lane & 15);
    const int kbaseA = (lane >> 4) * 8;
    const int colB   = lane & 15;
    const int kbaseB = (lane >> 4) * 16;

    v8f acc = {};

#pragma unroll 2
    for (int k = 0; k < K; k += 32) {
        v16h a, b;
        const float* xrow = X + (size_t)rowA * ldx + k;
#pragma unroll
        for (int i = 0; i < 16; ++i) {
            const int v = i >> 1, p = i & 1;
            const int kl = (v < 4) ? (kbaseA + 2 * v + p)
                                   : (16 + kbaseA + 2 * (v - 4) + p);
            a[i] = (_Float16)xrow[kl];
        }
#pragma unroll
        for (int i = 0; i < 16; ++i) {
            const int kl = k + ((i < 8) ? (kbaseB / 2 + i) : (16 + kbaseB / 2 + (i - 8)));
            b[i] = (_Float16)W[(size_t)kl * ldw + n0 + colB];
        }
        acc = __builtin_amdgcn_wmma_f32_16x16x32_f16(
                  false, a, false, b, (short)0, acc, false, false);
    }

    const int cn   = n0 + (lane & 15);
    const int mrow = m0 + (lane >> 4) * 8;
    const float bv = bias[cn];
#pragma unroll
    for (int j = 0; j < 8; ++j) {
        Y[(size_t)(mrow + j) * ldy + cn] = acc[j] + bv;
    }
}

// ---------------------------------------------------------------------------
// Paged attention decode, one workgroup per (batch, head).
// Phase 1: K streamed via double-buffered async global->LDS chunks
//          (ASYNCcnt + s_wait_asynccnt), scores -> LDS.
// Softmax: two-pass with tree reductions.
// Phase 2: weighted V sum, direct coalesced global stream (prefetched in P1).
// ---------------------------------------------------------------------------
__device__ __forceinline__
void issue_k_chunk(const float* __restrict__ kcache,
                   const float* __restrict__ vcache,
                   const int* __restrict__ sbt,
                   int n, int c0, float* dstbuf, int t)
{
#pragma unroll
    for (int j = 0; j < OPS_PER_CHUNK; ++j) {
        const int o     = j * 256 + t;          // 0..1023, 16B granules
        const int kc    = o >> 4;               // key within chunk (0..63)
        const int piece = o & 15;               // 16B piece within 256B key
        const int k     = c0 + kc;
        const int blk   = sbt[k >> 4];
        const size_t goff =
            (((size_t)blk * BLKSZ + (k & 15)) * NHEADS + n) * HEADDIM + piece * 4;
        const float* gp = kcache + goff;
        float*       lp = dstbuf + kc * HEADDIM + piece * 4;
#if HAVE_ASYNC_LDS
        __builtin_amdgcn_global_load_async_to_lds_b128(
            (gas_v4i_ptr)gp, (las_v4i_ptr)lp, 0, 0);
#else
        *(float4*)lp = *(const float4*)gp;
#endif
        // pull matching V cacheline toward L2 for phase 2 (one per 64B)
        if ((t & 3) == 0) __builtin_prefetch(vcache + goff, 0, 1);
    }
}

__global__ __launch_bounds__(256)
void paged_attn_decode(const float* __restrict__ q,        // [B, NH*HD]
                       const float* __restrict__ kcache,   // [NB, BS, NH, HD]
                       const float* __restrict__ vcache,   // [NB, BS, NH, HD]
                       const int*   __restrict__ block_tables, // [B, BLK_PER_SEQ]
                       const int*   __restrict__ seq_lens,     // [B]
                       float*       __restrict__ attn_out)     // [B, NH*HD]
{
    const int b    = blockIdx.x;
    const int n    = blockIdx.y;
    const int t    = threadIdx.x;   // 0..255
    const int w    = t >> 5;        // wave 0..7
    const int lane = t & 31;

    __shared__ float s_kbuf[2][CHUNK_FLTS];   // 2 x 16 KB ring
    __shared__ float s_scores[MAX_KV];        // 8 KB
    __shared__ int   s_bt[BLK_PER_SEQ];
    __shared__ float s_q[HEADDIM];
    __shared__ float s_red[256];
    __shared__ float s_out[4][HEADDIM];

    const int L  = seq_lens[b];
    const int np = (L + CHUNK_KEYS - 1) / CHUNK_KEYS;

    if (t < BLK_PER_SEQ) s_bt[t] = block_tables[b * BLK_PER_SEQ + t];
    if (t < HEADDIM)     s_q[t]  = q[(size_t)b * HID + n * HEADDIM + t];
    __syncthreads();

    // ---- Phase 1: scores via async-staged K chunks ----
    issue_k_chunk(kcache, vcache, s_bt, n, 0, s_kbuf[0], t);

    const float4 qv = *(const float4*)(s_q + (lane & 15) * 4);

    for (int p = 0; p < np; ++p) {
        if (p + 1 < np) {
            issue_k_chunk(kcache, vcache, s_bt, n,
                          (p + 1) * CHUNK_KEYS, s_kbuf[(p + 1) & 1], t);
            ASYNC_WAIT(4);   // chunk p complete, chunk p+1 in flight
        } else {
            ASYNC_WAIT(0);
        }
        __syncthreads();     // all waves' pieces of chunk p visible

        const float* kb = s_kbuf[p & 1];
        const int c0 = p * CHUNK_KEYS;
#pragma unroll
        for (int s = 0; s < 4; ++s) {
            const int kc = s * 16 + 2 * w + (lane >> 4);   // key pair per wave
            const int k  = c0 + kc;
            const float4 kv = ((const float4*)(kb + kc * HEADDIM))[lane & 15];
            float part = kv.x * qv.x + kv.y * qv.y + kv.z * qv.z + kv.w * qv.w;
            part += __shfl_xor(part, 8, 32);   // 16-lane reduce (stays in half)
            part += __shfl_xor(part, 4, 32);
            part += __shfl_xor(part, 2, 32);
            part += __shfl_xor(part, 1, 32);
            if ((lane & 15) == 0 && k < L) s_scores[k] = part * ATTN_SCALE;
        }
        __syncthreads();     // safe to overwrite buffer (p&1) next iteration
    }

    // ---- softmax: max reduce ----
    float m = -__builtin_inff();
    for (int k = t; k < L; k += 256) m = fmaxf(m, s_scores[k]);
    s_red[t] = m;
    __syncthreads();
    for (int off = 128; off > 0; off >>= 1) {
        if (t < off) s_red[t] = fmaxf(s_red[t], s_red[t + off]);
        __syncthreads();
    }
    m = s_red[0];
    __syncthreads();

    // ---- softmax: exp + sum reduce ----
    float sum = 0.f;
    for (int k = t; k < L; k += 256) {
        const float e = __expf(s_scores[k] - m);
        s_scores[k] = e;
        sum += e;
    }
    s_red[t] = sum;
    __syncthreads();
    for (int off = 128; off > 0; off >>= 1) {
        if (t < off) s_red[t] += s_red[t + off];
        __syncthreads();
    }
    const float inv = 1.f / s_red[0];

    // ---- Phase 2: out[d] = sum_k p[k] * V[k][d] (4 groups x 64 lanes) ----
    const int d = t & 63;
    const int g = t >> 6;
    float acc = 0.f;
    for (int k = g; k < L; k += 4) {
        const int blk = s_bt[k >> 4];
        const float* vp = vcache +
            (((size_t)blk * BLKSZ + (k & 15)) * NHEADS + n) * HEADDIM;
        acc += s_scores[k] * vp[d];
    }
    s_out[g][d] = acc;
    __syncthreads();
    if (t < HEADDIM) {
        const float o = (s_out[0][t] + s_out[1][t] + s_out[2][t] + s_out[3][t]) * inv;
        attn_out[(size_t)b * HID + n * HEADDIM + t] = o;
    }
}

// ---------------------------------------------------------------------------
// Launch: qproj (WMMA, only the Q slice of w_attn) -> attention -> outproj
// ---------------------------------------------------------------------------
extern "C" void kernel_launch(void* const* d_in, const int* in_sizes, int n_in,
                              void* d_out, int out_size, void* d_ws, size_t ws_size,
                              hipStream_t stream)
{
    const float* hidden       = (const float*)d_in[0];   // [32,1,1024]
    const float* kv_cache     = (const float*)d_in[1];   // [2,4096,16,16,64]
    const int*   block_tables = (const int*)  d_in[2];   // [32,128]
    const int*   seq_lens     = (const int*)  d_in[3];   // [32]
    const float* w_attn       = (const float*)d_in[4];   // [1024,3072]
    const float* b_attn       = (const float*)d_in[5];   // [3072]
    const float* w_proj       = (const float*)d_in[6];   // [1024,1024]
    const float* b_proj       = (const float*)d_in[7];   // [1024]
    float*       out          = (float*)d_out;           // [32,1,1024]

    const size_t kv_half = (size_t)NUM_BLOCKS * BLKSZ * NHEADS * HEADDIM;
    const float* kcache  = kv_cache;
    const float* vcache  = kv_cache + kv_half;

    float* q_ws    = (float*)d_ws;                       // [32,1024]
    float* attn_ws = q_ws + (size_t)BATCH * HID;         // [32,1024]

    // 1) Q projection: q = hidden @ w_attn[:, :1024] + b_attn[:1024]
    gemm_wmma_f16<<<dim3(HID / 16, BATCH / 16), 32, 0, stream>>>(
        hidden, HID, w_attn, 3 * HID, b_attn, q_ws, HID, HID);

    // 2) Paged attention decode (async K staging + softmax + V stream)
    paged_attn_decode<<<dim3(BATCH, NHEADS), 256, 0, stream>>>(
        q_ws, kcache, vcache, block_tables, seq_lens, attn_ws);

    // 3) Output projection: out = attn @ w_proj + b_proj
    gemm_wmma_f16<<<dim3(HID / 16, BATCH / 16), 32, 0, stream>>>(
        attn_ws, HID, w_proj, HID, b_proj, out, HID, HID);
}
